// LSTMNetwork_12068858102025
// MI455X (gfx1250) — matrix operand order repaired and checked by reading, not compiled, not run on hardware
//
#include <hip/hip_runtime.h>
#include <hip/hip_bf16.h>
#include <stddef.h>

// ---------------------------------------------------------------------------
// LSTM + temporal attention + MLP for MI455X (gfx1250, wave32, WMMA).
// Shapes: T=256, B=512, D=256, H=512, 4H=2048, K=D+H=768, F=64, O=1.
//
// Key idea this revision: all WMMA operands live in GLOBAL memory already in
// the exact per-lane fragment layout (ISA 7.12.2), so the GEMM inner loop is
// nothing but global_load_b128 -> v_wmma_f32_16x16x32_bf16. No LDS, no
// barriers, no fp32->bf16 conversion in the hot loop.
//   * wswz : [kb(24)][nt(128)][lane(32)][e(16)] bf16  B-fragments (3 MB, L2)
//   * xswz : [kb(8)][mt(32)][lane(32)][e(16)]  bf16  A-fragments of x_t
//   * hswz : [kb(16)][mt(32)][lane(32)][e(16)] bf16  A-fragments of h_{t-1}
// lstm_cell writes hswz for the next step and swizzles x_{t+1} with its spare
// threads; attention uses online softmax so hs(T,B,H) is never materialized.
// ---------------------------------------------------------------------------

typedef __bf16 bf16_t;
typedef __attribute__((ext_vector_type(16))) __bf16 v16bf;
typedef __attribute__((ext_vector_type(8)))  float  v8f;

#define TT 256
#define BB 512
#define DD 256
#define HH 512
#define NG 2048          // 4*H
#define KK 768           // D + H

// fragment-layout linear index (element units) for the A operand:
// value A[b][j] lives at ((kb*32 + mt)*32 + lane)*16 + e
// with kb=j>>5, krel=j&31, hf=(krel>>3)&1, e=(krel&7)+8*(krel>>4),
// lane=hf*16+(b&15), mt=b>>4.  (Inverse of kmap(e,hf)=e+8*hf+8*(e>>3).)
__device__ __forceinline__ size_t aswz_idx(int b, int j) {
  int kb   = j >> 5;
  int krel = j & 31;
  int hf   = (krel >> 3) & 1;
  int e    = (krel & 7) + 8 * (krel >> 4);
  int lane = hf * 16 + (b & 15);
  int mt   = b >> 4;
  return (((size_t)kb * 32 + mt) * 32 + lane) * 16 + e;
}

__device__ __forceinline__ unsigned pack2(float a, float b) {
  union { bf16_t h[2]; unsigned u; } p;
  p.h[0] = (bf16_t)a;
  p.h[1] = (bf16_t)b;
  return p.u;
}

__device__ __forceinline__ v8f wmma_bf16(v16bf a, v16bf b, v8f c) {
  return __builtin_amdgcn_wmma_f32_16x16x32_bf16(false, a, false, b,
                                                 (short)0, c, false, false);
}

// ---------------------------------------------------------------- setup -----
// B-fragment swizzle of [W_ih ; W_hh] -> wswz, plus fused bias.
// Element ((kb*128+nt)*32+lane)*16+e  <-  Wcat[k=kb*32+hf*8+(e&7)+16*(e>>3)][n=nt*16+l15].
__global__ void pack_weights(const float* __restrict__ Wih,
                             const float* __restrict__ Whh,
                             const float* __restrict__ bih,
                             const float* __restrict__ bhh,
                             bf16_t* __restrict__ wswz,
                             float* __restrict__ bias) {
  int p = blockIdx.x * blockDim.x + threadIdx.x;   // pair id, 0 .. KK*NG/2-1
  int el   = (2 * p) & 15;                         // even e
  int rest = (2 * p) >> 4;                         // (kb*128+nt)*32+lane
  int lane = rest & 31;
  int rest2 = rest >> 5;
  int nt = rest2 & 127;
  int kb = rest2 >> 7;
  int hf = lane >> 4, l15 = lane & 15;
  int k = kb * 32 + hf * 8 + (el & 7) + 16 * (el >> 3); // k, k+1 (same matrix)
  int n = nt * 16 + l15;
  float2 wv = (k < DD) ? *(const float2*)(Wih + (size_t)n * DD + k)
                       : *(const float2*)(Whh + (size_t)n * HH + (k - DD));
  *(unsigned*)(wswz + 2 * (size_t)p) = pack2(wv.x, wv.y);
  if (p < NG) bias[p] = bih[p] + bhh[p];
}

__global__ void init_state(const float* __restrict__ h0,
                           const float* __restrict__ c0,
                           const float* __restrict__ x0,
                           float* __restrict__ c, float* __restrict__ ctx,
                           bf16_t* __restrict__ hswz,
                           bf16_t* __restrict__ xswz,
                           float* __restrict__ m, float* __restrict__ s) {
  int idx = blockIdx.x * blockDim.x + threadIdx.x; // 0 .. B*H/2-1
  int b = idx >> 8;
  int j = (idx & 255) * 2;
  size_t lin = (size_t)b * HH + j;
  float2 hv = *(const float2*)(h0 + lin);
  float2 cv = *(const float2*)(c0 + lin);
  *(unsigned*)(hswz + aswz_idx(b, j)) = pack2(hv.x, hv.y);
  c[lin] = cv.x;  c[lin + 1] = cv.y;
  ctx[lin] = 0.f; ctx[lin + 1] = 0.f;
  if (idx < BB) { m[idx] = -3.0e38f; s[idx] = 0.f; }
  // swizzle x_0 (D=256: 65536 pairs)
  if (idx < (BB * DD) / 2) {
    int row = idx >> 7;
    int dp  = (idx & 127) * 2;
    float2 xv = *(const float2*)(x0 + (size_t)row * DD + dp);
    *(unsigned*)(xswz + aswz_idx(row, dp)) = pack2(xv.x, xv.y);
  }
}

// ------------------------------------------------------- WMMA gate GEMM -----
// gates(B x 4H) = [x_t | h_prev] @ Wcat + bias.  Pure register-fragment GEMM:
// global_load_b128 pairs feed v_wmma_f32_16x16x32_bf16 directly.
__global__ __launch_bounds__(256)
void gemm_gates(const bf16_t* __restrict__ xswz,
                const bf16_t* __restrict__ hswz,
                const bf16_t* __restrict__ wswz,
                const float* __restrict__ bias,
                float* __restrict__ gates) {
  const int tid  = threadIdx.x;
  const int lane = tid & 31;
  const int wave = tid >> 5;            // 8 waves
  const int wm   = wave >> 2;           // 0..1 : 64-row slab
  const int wn   = wave & 3;            // 0..3 : 32-col slab
  const int hf   = lane >> 4;
  const int l15  = lane & 15;
  const int mt0  = blockIdx.y * 8 + wm * 4;   // 4 M-tiles per wave
  const int nt0  = blockIdx.x * 8 + wn * 2;   // 2 N-tiles per wave

  v8f acc[4][2] = {};

  const bf16_t* wp = wswz + ((size_t)nt0 * 32 + lane) * 16;  // + kb*65536
  const bf16_t* xp = xswz + ((size_t)mt0 * 32 + lane) * 16;  // + kb*16384 + i*512
  const bf16_t* hp = hswz + ((size_t)mt0 * 32 + lane) * 16;

  // ---- x part: kb = 0..7 ----
#pragma unroll 2
  for (int kb = 0; kb < 8; ++kb) {
    __builtin_prefetch((const void*)(wp + (size_t)(kb + 1) * 65536), 0, 2);
    v16bf b0 = *(const v16bf*)(wp + (size_t)kb * 65536);
    v16bf b1 = *(const v16bf*)(wp + (size_t)kb * 65536 + 512);
#pragma unroll
    for (int i = 0; i < 4; ++i) {
      v16bf a = *(const v16bf*)(xp + (size_t)kb * 16384 + (size_t)i * 512);
      acc[i][0] = wmma_bf16(a, b0, acc[i][0]);
      acc[i][1] = wmma_bf16(a, b1, acc[i][1]);
    }
  }
  // ---- h part: kb = 8..23 ----
#pragma unroll 2
  for (int kb = 0; kb < 16; ++kb) {
    __builtin_prefetch((const void*)(wp + (size_t)(kb + 9) * 65536), 0, 2);
    v16bf b0 = *(const v16bf*)(wp + (size_t)(kb + 8) * 65536);
    v16bf b1 = *(const v16bf*)(wp + (size_t)(kb + 8) * 65536 + 512);
#pragma unroll
    for (int i = 0; i < 4; ++i) {
      v16bf a = *(const v16bf*)(hp + (size_t)kb * 16384 + (size_t)i * 512);
      acc[i][0] = wmma_bf16(a, b0, acc[i][0]);
      acc[i][1] = wmma_bf16(a, b1, acc[i][1]);
    }
  }

  // Epilogue: D layout (ISA 7.12.2): VGPR r -> row r + 8*hf, col = l15.
#pragma unroll
  for (int i = 0; i < 4; ++i) {
#pragma unroll
    for (int j = 0; j < 2; ++j) {
      int col = (nt0 + j) * 16 + l15;
      float bv = bias[col];
#pragma unroll
      for (int r = 0; r < 8; ++r) {
        int row = (mt0 + i) * 16 + r + 8 * hf;
        gates[(size_t)row * NG + col] = acc[i][j][r] + bv;
      }
    }
  }
}

// ------------------------------------------ LSTM cell + online attention ----
// One block per batch row b; thread handles (j, j+1), j = 2*tid.
// Also: blocks 0..255 swizzle x_{t+1} for the next step's GEMM.
__global__ __launch_bounds__(256)
void lstm_cell(const float* __restrict__ gates,
               float* __restrict__ c,
               float* __restrict__ hfp32,
               bf16_t* __restrict__ hswz,
               float* __restrict__ ctx,
               float* __restrict__ mrun,
               float* __restrict__ srun,
               const float* __restrict__ attn_w,
               const float* __restrict__ attn_b,
               const float* __restrict__ xnext,   // may be nullptr (last step)
               bf16_t* __restrict__ xswz) {
  const int b   = blockIdx.x;
  const int tid = threadIdx.x;
  const int j   = 2 * tid;
  __shared__ float red[8];
  __shared__ float ab[2];

  // spare work: swizzle next step's x slab (65536 pairs over first 256 blocks)
  if (xnext != nullptr && b < 256) {
    int p   = b * 256 + tid;
    int row = p >> 7;
    int dp  = (p & 127) * 2;
    float2 xv = *(const float2*)(xnext + (size_t)row * DD + dp);
    *(unsigned*)(xswz + aswz_idx(row, dp)) = pack2(xv.x, xv.y);
  }

  const float* grow = gates + (size_t)b * NG;
  float hv[2];
#pragma unroll
  for (int u = 0; u < 2; ++u) {
    int jj = j + u;
    float gi = grow[jj];
    float gf = grow[jj + 512];
    float gg = grow[jj + 1024];
    float go = grow[jj + 1536];
    float si = 1.f / (1.f + __expf(-gi));
    float sf = 1.f / (1.f + __expf(-gf));
    float so = 1.f / (1.f + __expf(-go));
    size_t lin = (size_t)b * HH + jj;
    float cn = sf * c[lin] + si * tanhf(gg);
    c[lin] = cn;
    float hn = so * tanhf(cn);
    hfp32[lin] = hn;
    hv[u] = hn;
  }
  // h -> bf16 A-fragment layout for next step's GEMM (one 32-bit store)
  *(unsigned*)(hswz + aswz_idx(b, j)) = pack2(hv[0], hv[1]);

  // attention score: dot(h_row, attn_w), wave32 shuffle + cross-wave LDS
  float part = hv[0] * attn_w[j] + hv[1] * attn_w[j + 1];
#pragma unroll
  for (int off = 16; off > 0; off >>= 1) part += __shfl_down(part, off, 32);
  if ((tid & 31) == 0) red[tid >> 5] = part;
  __syncthreads();
  if (tid == 0) {
    float sc = attn_b[0];
#pragma unroll
    for (int w = 0; w < 8; ++w) sc += red[w];
    float mo = mrun[b], so = srun[b];
    float nm = fmaxf(mo, sc);
    float alpha = __expf(mo - nm);
    float beta  = __expf(sc - nm);
    srun[b] = so * alpha + beta;
    mrun[b] = nm;
    ab[0] = alpha; ab[1] = beta;
  }
  __syncthreads();
  float alpha = ab[0], beta = ab[1];
#pragma unroll
  for (int u = 0; u < 2; ++u) {
    size_t lin = (size_t)b * HH + j + u;
    ctx[lin] = ctx[lin] * alpha + hv[u] * beta;
  }
}

// ---------------------------------------------------------- final MLP -------
__global__ __launch_bounds__(64)
void final_mlp(const float* __restrict__ ctx, const float* __restrict__ srun,
               const float* __restrict__ W1, const float* __restrict__ b1,
               const float* __restrict__ W2, const float* __restrict__ b2,
               const float* __restrict__ W3, const float* __restrict__ b3,
               const float* __restrict__ W4, const float* __restrict__ b4,
               float* __restrict__ out) {
  const int b = blockIdx.x;
  const int tid = threadIdx.x;
  __shared__ float crow[HH];
  __shared__ float x1[64];
  __shared__ float x2[64];
  __shared__ float x3[32];
  float inv = 1.f / srun[b];
#pragma unroll
  for (int u = 0; u < 8; ++u)
    crow[tid + u * 64] = ctx[(size_t)b * HH + tid + u * 64] * inv;
  __syncthreads();
  float a = b1[tid];
  for (int j = 0; j < HH; ++j) a += crow[j] * W1[tid * HH + j];
  x1[tid] = tanhf(a);
  __syncthreads();
  a = b2[tid];
  for (int j = 0; j < 64; ++j) a += x1[j] * W2[tid * 64 + j];
  x2[tid] = tanhf(a);
  __syncthreads();
  if (tid < 32) {
    a = b3[tid];
    for (int j = 0; j < 64; ++j) a += x2[j] * W3[tid * 64 + j];
    x3[tid] = tanhf(a);
  }
  __syncthreads();
  if (tid == 0) {
    a = b4[0];
    for (int j = 0; j < 32; ++j) a += x3[j] * W4[j];
    out[b] = a;
  }
}

__global__ void copy_hc(const float* __restrict__ h, const float* __restrict__ c,
                        float* __restrict__ oh, float* __restrict__ oc) {
  int i = blockIdx.x * blockDim.x + threadIdx.x;
  oh[i] = h[i];
  oc[i] = c[i];
}

// ------------------------------------------------------------ launcher ------
extern "C" void kernel_launch(void* const* d_in, const int* in_sizes, int n_in,
                              void* d_out, int out_size, void* d_ws, size_t ws_size,
                              hipStream_t stream) {
  const float* inputs = (const float*)d_in[0];   // (T,B,D)
  const float* h0     = (const float*)d_in[1];
  const float* c0     = (const float*)d_in[2];
  const float* Wih    = (const float*)d_in[3];   // (4H,D)
  const float* Whh    = (const float*)d_in[4];   // (4H,H)
  const float* bih    = (const float*)d_in[5];
  const float* bhh    = (const float*)d_in[6];
  const float* attn_w = (const float*)d_in[7];
  const float* attn_b = (const float*)d_in[8];
  const float* W1 = (const float*)d_in[9];
  const float* b1 = (const float*)d_in[10];
  const float* W2 = (const float*)d_in[11];
  const float* b2 = (const float*)d_in[12];
  const float* W3 = (const float*)d_in[13];
  const float* b3 = (const float*)d_in[14];
  const float* W4 = (const float*)d_in[15];
  const float* b4 = (const float*)d_in[16];

  char* base = (char*)d_ws;
  size_t off = 0;
  auto alloc = [&](size_t bytes) {
    void* p = base + off;
    off += (bytes + 255) & ~(size_t)255;
    return p;
  };
  bf16_t* wswz = (bf16_t*)alloc((size_t)KK * NG * sizeof(bf16_t));  // 3 MB
  float*  bias = (float*) alloc((size_t)NG * sizeof(float));
  float*  gates= (float*) alloc((size_t)BB * NG * sizeof(float));   // 4 MB
  bf16_t* hswz = (bf16_t*)alloc((size_t)BB * HH * sizeof(bf16_t));  // 512 KB
  bf16_t* xswz = (bf16_t*)alloc((size_t)BB * DD * sizeof(bf16_t));  // 256 KB
  float*  hfp32= (float*) alloc((size_t)BB * HH * sizeof(float));
  float*  cbuf = (float*) alloc((size_t)BB * HH * sizeof(float));
  float*  ctx  = (float*) alloc((size_t)BB * HH * sizeof(float));
  float*  mrun = (float*) alloc((size_t)BB * sizeof(float));
  float*  srun = (float*) alloc((size_t)BB * sizeof(float));

  float* out_pred = (float*)d_out;              // (B, 1)
  float* out_h    = out_pred + (size_t)BB;      // (B, H)
  float* out_c    = out_h    + (size_t)BB * HH; // (B, H)

  pack_weights<<<(KK * NG / 2) / 256, 256, 0, stream>>>(Wih, Whh, bih, bhh,
                                                        wswz, bias);
  init_state<<<(BB * HH / 2) / 256, 256, 0, stream>>>(h0, c0, inputs, cbuf, ctx,
                                                      hswz, xswz, mrun, srun);

  dim3 ggrid(NG / 128, BB / 128);               // 16 x 4
  for (int t = 0; t < TT; ++t) {
    gemm_gates<<<ggrid, 256, 0, stream>>>(xswz, hswz, wswz, bias, gates);
    const float* xnext = (t + 1 < TT) ? inputs + (size_t)(t + 1) * BB * DD
                                      : nullptr;
    lstm_cell<<<BB, 256, 0, stream>>>(gates, cbuf, hfp32, hswz, ctx,
                                      mrun, srun, attn_w, attn_b, xnext, xswz);
  }
  final_mlp<<<BB, 64, 0, stream>>>(ctx, srun, W1, b1, W2, b2, W3, b3, W4, b4,
                                   out_pred);
  copy_hc<<<(BB * HH) / 256, 256, 0, stream>>>(hfp32, cbuf, out_h, out_c);
}
